// MPCTC_15504831938820
// MI455X (gfx1250) — compile-verified, matrix operand order checked by reading, not run on hardware
//
#include <hip/hip_runtime.h>
#include <stdint.h>
#include <stddef.h>

// ---------------- problem constants ----------------
constexpr int T_  = 256;
constexpr int B_  = 64;
constexpr int D_  = 512;
constexpr int V_  = 80;
constexpr int H_  = 8;
constexpr int FF_ = 2048;
constexpr int TB_ = T_ * B_;
constexpr int BLANK_ = V_ - 1;
constexpr int MASK_  = V_ - 2;

// ---------------- CDNA5 WMMA types ----------------
typedef __attribute__((ext_vector_type(16))) __bf16 v16bf;
typedef __attribute__((ext_vector_type(8)))  __bf16 v8bf;
typedef __attribute__((ext_vector_type(8)))  float  v8f;

__device__ __forceinline__ __bf16 f2bf(float f) {
    unsigned u = __builtin_bit_cast(unsigned, f);
    unsigned r = (u + 0x7FFFu + ((u >> 16) & 1u)) >> 16;
    return __builtin_bit_cast(__bf16, (unsigned short)r);
}

__device__ __forceinline__ v8f wmma_bf16(v16bf a, v16bf b, v8f c) {
    return __builtin_amdgcn_wmma_f32_16x16x32_bf16(false, a, false, b,
                                                   (short)0, c, false, false);
}

// Load a 16x32 bf16 A/B fragment slice for one lane: two contiguous 16B loads.
// Per ISA 7.12.2: lanes 0-15 take K {0..7,16..23}+base, lanes 16-31 take {8..15,24..31}.
// Caller passes p already offset by (lane>>4)*8 halves.
__device__ __forceinline__ v16bf load_frag(const __bf16* p) {
    v8bf lo = *(const v8bf*)p;
    v8bf hi = *(const v8bf*)(p + 16);
    v16bf r;
#pragma unroll
    for (int i = 0; i < 8; ++i) { r[i] = lo[i]; r[i + 8] = hi[i]; }
    return r;
}

// ============================================================================
// Main GEMM:  Out = A(MxK) * W(NxK)^T + bias
// Block tile 64x64: 4 waves x (16 rows x 64 cols). The W panel (64 cols x 32 k,
// 4KB bf16) is staged into LDS with CDNA5 async global->LDS DMA
// (global_load_async_to_lds_b128, ASYNCcnt) and shared by all 4 waves,
// double-buffered so the next panel's fill overlaps the current panel's WMMAs.
// MODE 0: f32 out + bias ; MODE 1: bf16 out + bias ; MODE 2: bf16 + bias + relu
// ============================================================================
template <int MODE>
__global__ __launch_bounds__(128) void gemm_bf16_async(
    const __bf16* __restrict__ A, const __bf16* __restrict__ W,
    const float* __restrict__ bias, void* __restrict__ Out,
    int M, int N, int K)
{
    // [buf][col][k] ; 64B per col-row, 4096B per buffer; array sits at LDS offset 0
    __shared__ __bf16 lds_w[2][64][32];

    const int lane = threadIdx.x & 31;
    const int wave = threadIdx.x >> 5;
    const int lr = lane & 15, hb = lane >> 4;
    const int m0 = blockIdx.x * 64 + wave * 16;
    const int n0 = blockIdx.y * 64;

    // async-stage one 64x32 W panel: 256 x 16B chunks over 128 threads
    auto stage = [&](int buf, int k) {
#pragma unroll
        for (int it = 0; it < 2; ++it) {
            const int ch  = threadIdx.x + it * 128;       // 0..255
            const int row = ch >> 2, q = ch & 3;
            const __bf16* src = W + (size_t)(n0 + row) * K + k + q * 8;
            const unsigned dst = (unsigned)(buf * 4096 + row * 64 + q * 16);
            asm volatile("global_load_async_to_lds_b128 %0, %1, off"
                         :: "v"(dst), "v"((unsigned long long)(uintptr_t)src)
                         : "memory");
        }
    };

    v8f acc[4] = {};
    const __bf16* arow = A + (size_t)(m0 + lr) * K + hb * 8;
    const int KT = K / 32;

    stage(0, 0);
    for (int kk = 0; kk < KT; ++kk) {
        const int buf = kk & 1;
        if (kk + 1 < KT) {
            stage(buf ^ 1, (kk + 1) * 32);
            asm volatile("s_wait_asynccnt 0x2" ::: "memory"); // prev panel landed
        } else {
            asm volatile("s_wait_asynccnt 0x0" ::: "memory");
        }
        __syncthreads();

        __builtin_prefetch(arow + kk * 32 + 128, 0, 1);
        v16bf af = load_frag(arow + kk * 32);
#pragma unroll
        for (int nn = 0; nn < 4; ++nn) {
            v16bf bf = load_frag(&lds_w[buf][nn * 16 + lr][hb * 8]); // ds_load_b128 x2
            acc[nn] = wmma_bf16(af, bf, acc[nn]);
        }
        __syncthreads(); // all waves done with buf before it is refilled
    }

#pragma unroll
    for (int nn = 0; nn < 4; ++nn) {
        const int col = n0 + nn * 16 + lr;
        const float bv = bias[col];
#pragma unroll
        for (int r = 0; r < 8; ++r) {
            const int row = m0 + r + hb * 8;   // C layout: M = r + 8*(lane>=16)
            float v = acc[nn][r] + bv;
            if (MODE == 2) v = v > 0.f ? v : 0.f;
            if (MODE == 0) ((float*)Out)[(size_t)row * N + col] = v;
            else           ((__bf16*)Out)[(size_t)row * N + col] = f2bf(v);
        }
    }
}

// ---------------- small direct-load GEMM for N=80 output projection ---------
template <int NTW, int MODE>
__global__ __launch_bounds__(128) void gemm_bf16_wmma(
    const __bf16* __restrict__ A, const __bf16* __restrict__ W,
    const float* __restrict__ bias, void* __restrict__ Out,
    int M, int N, int K)
{
    const int lane = threadIdx.x & 31;
    const int wave = threadIdx.x >> 5;
    const int gw   = blockIdx.x * 4 + wave;
    const int nGrp = N / (16 * NTW);
    const int mi   = gw / nGrp;
    const int ni   = gw % nGrp;
    if (mi * 16 >= M) return;
    const int lr = lane & 15, hb = lane >> 4;
    const int m0 = mi * 16, n0 = ni * 16 * NTW;

    v8f acc[NTW] = {};

    const __bf16* arow = A + (size_t)(m0 + lr) * K + hb * 8;
    for (int k = 0; k < K; k += 32) {
        __builtin_prefetch(arow + k + 128, 0, 1);
        v16bf af = load_frag(arow + k);
#pragma unroll
        for (int nn = 0; nn < NTW; ++nn) {
            const __bf16* wrow = W + (size_t)(n0 + nn * 16 + lr) * K + k + hb * 8;
            v16bf bf = load_frag(wrow);
            acc[nn] = wmma_bf16(af, bf, acc[nn]);
        }
    }

#pragma unroll
    for (int nn = 0; nn < NTW; ++nn) {
        const int col = n0 + nn * 16 + lr;
        const float bv = bias[col];
#pragma unroll
        for (int r = 0; r < 8; ++r) {
            const int row = m0 + r + hb * 8;
            float v = acc[nn][r] + bv;
            if (MODE == 2) v = v > 0.f ? v : 0.f;
            if (MODE == 0) ((float*)Out)[(size_t)row * N + col] = v;
            else           ((__bf16*)Out)[(size_t)row * N + col] = f2bf(v);
        }
    }
}

// ---------------- fused attention: one wave per (q-tile, head, batch) --------
// Q,K,V are (T*B, D) bf16 with head h in columns [h*64, h*64+64). O may alias Q.
__global__ __launch_bounds__(32) void attn_wmma(
    const __bf16* __restrict__ Q, const __bf16* __restrict__ Km,
    const __bf16* __restrict__ Vm, __bf16* __restrict__ O)
{
    __shared__ __bf16 lds_p[16][272];   // softmaxed P tile, skewed
    __shared__ __bf16 lds_v[256][72];   // V head, 144B row stride (16B aligned)
    const int lane = threadIdx.x;
    const int lr = lane & 15, hb = lane >> 4;
    const int qt = blockIdx.x, h = blockIdx.y, b = blockIdx.z;

    // stage V head (256 keys x 64 feats) into LDS with 16B vector copies
    for (int idx = lane; idx < 256 * 8; idx += 32) {
        const int key = idx >> 3, c = idx & 7;
        *(v8bf*)&lds_v[key][c * 8] =
            *(const v8bf*)(Vm + ((size_t)key * B_ + b) * D_ + h * 64 + c * 8);
    }

    // Q fragments (16 queries x dh=64 -> 2 K-steps)
    const __bf16* qrow = Q + ((size_t)(qt * 16 + lr) * B_ + b) * D_ + h * 64 + hb * 8;
    const v16bf aq0 = load_frag(qrow);
    const v16bf aq1 = load_frag(qrow + 32);

    // S = Q * K^T : 16 key-tiles x 2 WMMAs
    v8f s[16];
#pragma unroll
    for (int kt = 0; kt < 16; ++kt) {
        v8f acc = {};
        const __bf16* krow =
            Km + ((size_t)(kt * 16 + lr) * B_ + b) * D_ + h * 64 + hb * 8;
        acc = wmma_bf16(aq0, load_frag(krow), acc);
        acc = wmma_bf16(aq1, load_frag(krow + 32), acc);
        s[kt] = acc;
    }

    // row softmax (rows live in VGPR index r, N striped over 16-lane groups)
    const float scale = 0.125f;   // dh^-0.5
#pragma unroll
    for (int r = 0; r < 8; ++r) {
        float mx = -3.4e38f;
#pragma unroll
        for (int kt = 0; kt < 16; ++kt) { s[kt][r] *= scale; mx = fmaxf(mx, s[kt][r]); }
#pragma unroll
        for (int m = 1; m < 16; m <<= 1) mx = fmaxf(mx, __shfl_xor(mx, m, 32));
        float sum = 0.f;
#pragma unroll
        for (int kt = 0; kt < 16; ++kt) { float e = __expf(s[kt][r] - mx); s[kt][r] = e; sum += e; }
#pragma unroll
        for (int m = 1; m < 16; m <<= 1) sum += __shfl_xor(sum, m, 32);
        const float inv = 1.f / sum;
        const int row = r + hb * 8;
#pragma unroll
        for (int kt = 0; kt < 16; ++kt) lds_p[row][kt * 16 + lr] = f2bf(s[kt][r] * inv);
    }
    asm volatile("s_wait_dscnt 0" ::: "memory");

    // O = P * V : K=256 in 8 steps of 32
    v8f oacc[4] = {};
    for (int kk = 0; kk < 8; ++kk) {
        const int k0 = kk * 32 + hb * 8;
        v16bf ap;
#pragma unroll
        for (int i = 0; i < 8; ++i) { ap[i] = lds_p[lr][k0 + i]; ap[i + 8] = lds_p[lr][k0 + 16 + i]; }
#pragma unroll
        for (int nn = 0; nn < 4; ++nn) {
            const int n = nn * 16 + lr;
            v16bf bp;
#pragma unroll
            for (int i = 0; i < 8; ++i) { bp[i] = lds_v[k0 + i][n]; bp[i + 8] = lds_v[k0 + 16 + i][n]; }
            oacc[nn] = wmma_bf16(ap, bp, oacc[nn]);
        }
    }
#pragma unroll
    for (int nn = 0; nn < 4; ++nn)
#pragma unroll
        for (int r = 0; r < 8; ++r)
            O[((size_t)(qt * 16 + r + hb * 8) * B_ + b) * D_ + h * 64 + nn * 16 + lr] =
                f2bf(oacc[nn][r]);
}

// ---------------- CTC group preprocessing: one thread per batch column ------
__global__ void ctc_prep(const float* __restrict__ y, int* __restrict__ out_ids,
                         int* __restrict__ copy_src, int* __restrict__ pred_g,
                         float* __restrict__ gmax_g, int* __restrict__ gstart_g)
{
    const int b = threadIdx.x;
    if (b >= B_) return;
    int gid = -1, prev = -1;
    for (int t = 0; t < T_; ++t) {
        const float* yp = y + ((size_t)t * B_ + b) * V_;
        float mx = yp[0]; int am = 0;
        for (int v = 1; v < V_; ++v) { float val = yp[v]; if (val > mx) { mx = val; am = v; } }
        float se = 0.f;
        for (int v = 0; v < V_; ++v) se += __expf(yp[v] - mx);
        const float prob = 1.f / se;      // exp(max log_softmax)
        if (t == 0 || am != prev) {
            ++gid;
            pred_g[gid * B_ + b]   = am;
            gmax_g[gid * B_ + b]   = prob;
            gstart_g[gid * B_ + b] = t;
        } else if (prob > gmax_g[gid * B_ + b]) {
            gmax_g[gid * B_ + b] = prob;
        }
        prev = am;
    }
    const int ng = gid + 1;
    for (int t = 0; t < T_; ++t) out_ids[t * B_ + b] = BLANK_;
    int rank = 0;
    for (int g = 0; g < ng; ++g) {
        const int p = pred_g[g * B_ + b];
        if (p != BLANK_) { out_ids[rank * B_ + b] = p; ++rank; }
    }
    for (int t = 0; t < T_; ++t) {
        int cs = -1;
        if (t < ng) {
            if (gmax_g[t * B_ + b] >= 0.9f) cs = gstart_g[t * B_ + b];
            else out_ids[t * B_ + b] = MASK_;   // MASK overrides rank writes
        }
        copy_src[t * B_ + b] = cs;
    }
}

// ---------------- embedding gather -> f32 master + bf16 activation ----------
__global__ void embed_k(const int* __restrict__ ids, const float* __restrict__ emb,
                        float* __restrict__ hf, __bf16* __restrict__ hbf)
{
    const int row = blockIdx.x;            // t*B+b
    const int id = ids[row];
    const float* e = emb + (size_t)id * D_;
    for (int d = threadIdx.x; d < D_; d += blockDim.x) {
        const float v = e[d];
        hf[(size_t)row * D_ + d]  = v;
        hbf[(size_t)row * D_ + d] = f2bf(v);
    }
}

// ---------------- residual + LayerNorm (block = one token, D=512) ----------
__global__ __launch_bounds__(256) void resid_ln(
    const float* __restrict__ proj, float* __restrict__ hf,
    const float* __restrict__ g, const float* __restrict__ bt,
    __bf16* __restrict__ hbf, int withResid)
{
    __shared__ float red[256];
    const int row = blockIdx.x, tid = threadIdx.x;
    float v0 = hf[(size_t)row * D_ + tid];
    float v1 = hf[(size_t)row * D_ + tid + 256];
    if (withResid) {
        v0 += proj[(size_t)row * D_ + tid];
        v1 += proj[(size_t)row * D_ + tid + 256];
    }
    red[tid] = v0 + v1; __syncthreads();
    for (int st = 128; st > 0; st >>= 1) { if (tid < st) red[tid] += red[tid + st]; __syncthreads(); }
    const float mean = red[0] * (1.f / D_);
    __syncthreads();
    const float d0 = v0 - mean, d1 = v1 - mean;
    red[tid] = d0 * d0 + d1 * d1; __syncthreads();
    for (int st = 128; st > 0; st >>= 1) { if (tid < st) red[tid] += red[tid + st]; __syncthreads(); }
    const float inv = rsqrtf(red[0] * (1.f / D_) + 1e-5f);
    const float o0 = d0 * inv * g[tid]       + bt[tid];
    const float o1 = d1 * inv * g[tid + 256] + bt[tid + 256];
    hf[(size_t)row * D_ + tid]        = o0;
    hf[(size_t)row * D_ + tid + 256]  = o1;
    hbf[(size_t)row * D_ + tid]       = f2bf(o0);
    hbf[(size_t)row * D_ + tid + 256] = f2bf(o1);
}

// ---------------- f32 -> bf16 conversion -------------------------------
__global__ void cvt_bf16(const float* __restrict__ src, __bf16* __restrict__ dst, int n)
{
    const int i = blockIdx.x * blockDim.x + threadIdx.x;
    if (i < n) dst[i] = f2bf(src[i]);
}

// ---------------- final select: logits vs y[gstart] rows --------------------
__global__ void finalize_k(const float* __restrict__ logits, const float* __restrict__ y,
                           const int* __restrict__ copy_src, float* __restrict__ out)
{
    const int row = blockIdx.x;            // t*B+b
    const int b = row % B_;
    const int cs = copy_src[row];
    const float* src = (cs >= 0) ? (y + ((size_t)cs * B_ + b) * V_)
                                 : (logits + (size_t)row * V_);
    for (int v = threadIdx.x; v < V_; v += blockDim.x)
        out[(size_t)row * V_ + v] = src[v];
}

// ---------------- host side -------------------------------------------------
template <int MODE>
static void run_gemm64(const __bf16* A, const __bf16* W, const float* bias, void* out,
                       int M, int N, int K, hipStream_t s)
{
    dim3 grid(M / 64, N / 64);
    gemm_bf16_async<MODE><<<grid, 128, 0, s>>>(A, W, bias, out, M, N, K);
}

struct AttnPtr { const float *wq, *wk, *wv, *wo, *bq, *bk, *bv, *bo; };
struct LayerPtr {
    AttnPtr sa, ca;
    const float *ln1g, *ln1b, *ln2g, *ln2b, *ln3g, *ln3b, *w1, *b1, *w2, *b2;
};

extern "C" void kernel_launch(void* const* d_in, const int* in_sizes, int n_in,
                              void* d_out, int out_size, void* d_ws, size_t ws_size,
                              hipStream_t stream)
{
    (void)in_sizes; (void)n_in; (void)out_size; (void)ws_size;
    const float* x = (const float*)d_in[0];
    const float* y = (const float*)d_in[1];
    int pi = 2;
    const float* emb = (const float*)d_in[pi++];
    LayerPtr lp[2];
    for (int l = 0; l < 2; ++l) {
        AttnPtr* aps[2] = { &lp[l].sa, &lp[l].ca };
        for (int a = 0; a < 2; ++a) {
            aps[a]->wq = (const float*)d_in[pi++];
            aps[a]->wk = (const float*)d_in[pi++];
            aps[a]->wv = (const float*)d_in[pi++];
            aps[a]->wo = (const float*)d_in[pi++];
            aps[a]->bq = (const float*)d_in[pi++];
            aps[a]->bk = (const float*)d_in[pi++];
            aps[a]->bv = (const float*)d_in[pi++];
            aps[a]->bo = (const float*)d_in[pi++];
        }
        lp[l].ln1g = (const float*)d_in[pi++]; lp[l].ln1b = (const float*)d_in[pi++];
        lp[l].ln2g = (const float*)d_in[pi++]; lp[l].ln2b = (const float*)d_in[pi++];
        lp[l].ln3g = (const float*)d_in[pi++]; lp[l].ln3b = (const float*)d_in[pi++];
        lp[l].w1 = (const float*)d_in[pi++];   lp[l].b1 = (const float*)d_in[pi++];
        lp[l].w2 = (const float*)d_in[pi++];   lp[l].b2 = (const float*)d_in[pi++];
    }
    const float* lnfg = (const float*)d_in[pi++];
    const float* lnfb = (const float*)d_in[pi++];
    const float* ow   = (const float*)d_in[pi++];
    const float* ob   = (const float*)d_in[pi++];

    // ---- workspace layout ----
    char* ws = (char*)d_ws;
    size_t off = 0;
    auto take = [&](size_t bytes) -> char* {
        char* p = ws + off;
        off = (off + bytes + 255) & ~(size_t)255;
        return p;
    };
    int*    out_ids  = (int*)take((size_t)TB_ * 4);
    int*    copy_src = (int*)take((size_t)TB_ * 4);
    int*    pred_g   = (int*)take((size_t)TB_ * 4);
    float*  gmax_g   = (float*)take((size_t)TB_ * 4);
    int*    gstart_g = (int*)take((size_t)TB_ * 4);
    float*  hf     = (float*)take((size_t)TB_ * D_ * 4);
    __bf16* hbf    = (__bf16*)take((size_t)TB_ * D_ * 2);
    __bf16* xbf    = (__bf16*)take((size_t)TB_ * D_ * 2);
    __bf16* qbf    = (__bf16*)take((size_t)TB_ * D_ * 2);   // also holds attn output
    __bf16* kbf    = (__bf16*)take((size_t)TB_ * D_ * 2);
    __bf16* vbf    = (__bf16*)take((size_t)TB_ * D_ * 2);
    float*  proj   = (float*)take((size_t)TB_ * D_ * 4);
    __bf16* ffbf   = (__bf16*)take((size_t)TB_ * FF_ * 2);
    float*  logits = (float*)take((size_t)TB_ * V_ * 4);
    __bf16* wbf[2][10];
    for (int l = 0; l < 2; ++l) {
        for (int i = 0; i < 8; ++i) wbf[l][i] = (__bf16*)take((size_t)D_ * D_ * 2);
        wbf[l][8] = (__bf16*)take((size_t)FF_ * D_ * 2);   // w1
        wbf[l][9] = (__bf16*)take((size_t)D_ * FF_ * 2);   // w2
    }
    __bf16* owbf = (__bf16*)take((size_t)V_ * D_ * 2);

    auto cvt = [&](const float* s, __bf16* d, int n) {
        cvt_bf16<<<(n + 255) / 256, 256, 0, stream>>>(s, d, n);
    };

    // ---- weight + input conversion (f32 -> bf16) ----
    cvt(x, xbf, TB_ * D_);
    for (int l = 0; l < 2; ++l) {
        const AttnPtr* a2[2] = { &lp[l].sa, &lp[l].ca };
        for (int a = 0; a < 2; ++a) {
            cvt(a2[a]->wq, wbf[l][a * 4 + 0], D_ * D_);
            cvt(a2[a]->wk, wbf[l][a * 4 + 1], D_ * D_);
            cvt(a2[a]->wv, wbf[l][a * 4 + 2], D_ * D_);
            cvt(a2[a]->wo, wbf[l][a * 4 + 3], D_ * D_);
        }
        cvt(lp[l].w1, wbf[l][8], FF_ * D_);
        cvt(lp[l].w2, wbf[l][9], D_ * FF_);
    }
    cvt(ow, owbf, V_ * D_);

    // ---- CTC preprocessing + embedding ----
    ctc_prep<<<1, 64, 0, stream>>>(y, out_ids, copy_src, pred_g, gmax_g, gstart_g);
    embed_k<<<TB_, 128, 0, stream>>>(out_ids, emb, hf, hbf);

    // ---- decoder layers ----
    const dim3 attn_grid(T_ / 16, H_, B_);
    for (int l = 0; l < 2; ++l) {
        // self-attention
        run_gemm64<1>(hbf, wbf[l][0], lp[l].sa.bq, qbf, TB_, D_, D_, stream);
        run_gemm64<1>(hbf, wbf[l][1], lp[l].sa.bk, kbf, TB_, D_, D_, stream);
        run_gemm64<1>(hbf, wbf[l][2], lp[l].sa.bv, vbf, TB_, D_, D_, stream);
        attn_wmma<<<attn_grid, 32, 0, stream>>>(qbf, kbf, vbf, qbf);
        run_gemm64<0>(qbf, wbf[l][3], lp[l].sa.bo, proj, TB_, D_, D_, stream);
        resid_ln<<<TB_, 256, 0, stream>>>(proj, hf, lp[l].ln1g, lp[l].ln1b, hbf, 1);
        // cross-attention (K,V from x)
        run_gemm64<1>(hbf, wbf[l][4], lp[l].ca.bq, qbf, TB_, D_, D_, stream);
        run_gemm64<1>(xbf, wbf[l][5], lp[l].ca.bk, kbf, TB_, D_, D_, stream);
        run_gemm64<1>(xbf, wbf[l][6], lp[l].ca.bv, vbf, TB_, D_, D_, stream);
        attn_wmma<<<attn_grid, 32, 0, stream>>>(qbf, kbf, vbf, qbf);
        run_gemm64<0>(qbf, wbf[l][7], lp[l].ca.bo, proj, TB_, D_, D_, stream);
        resid_ln<<<TB_, 256, 0, stream>>>(proj, hf, lp[l].ln2g, lp[l].ln2b, hbf, 1);
        // feed-forward
        run_gemm64<2>(hbf, wbf[l][8], lp[l].b1, ffbf, TB_, FF_, D_, stream);
        run_gemm64<0>(ffbf, wbf[l][9], lp[l].b2, proj, TB_, D_, FF_, stream);
        resid_ln<<<TB_, 256, 0, stream>>>(proj, hf, lp[l].ln3g, lp[l].ln3b, hbf, 1);
    }

    // ---- final LN + output projection (N=80 = 5 x 16 tiles per wave) ----
    resid_ln<<<TB_, 256, 0, stream>>>(proj, hf, lnfg, lnfb, hbf, 0);
    {
        const int waves = (TB_ / 16) * (V_ / 80);
        gemm_bf16_wmma<5, 0><<<(waves + 3) / 4, 128, 0, stream>>>(
            hbf, owbf, ob, logits, TB_, V_, D_);
    }

    // ---- confident-group overwrite ----
    finalize_k<<<TB_, 80, 0, stream>>>(logits, y, copy_src, (float*)d_out);
}
